// QuantLinear_24721831756578
// MI455X (gfx1250) — matrix operand order repaired and checked by reading, not compiled, not run on hardware
//
#include <hip/hip_runtime.h>

typedef _Float16 v16h __attribute__((ext_vector_type(16)));
typedef _Float16 v8h  __attribute__((ext_vector_type(8)));
typedef _Float16 v2h  __attribute__((ext_vector_type(2)));
typedef float    v8f  __attribute__((ext_vector_type(8)));
typedef int      v8i  __attribute__((ext_vector_type(8)));

#define OUT_DIM 11008
#define IN_DIM  4096
#define BATCH   64
#define XROW    272          // 256 f16 + 16 pad -> 544B row stride (32B aligned)
#define N_TILES (OUT_DIM / 16)   // 688
#define WAVES_PER_BLOCK 8
#define GRID_X  (N_TILES / WAVES_PER_BLOCK)  // 86
#define KSPLIT  4
#define OUT_ELEMS (BATCH * OUT_DIM)          // 704512

// One wave computes a 64(M) x 16(N) partial tile over chunks_per_split*256 of K.
// 8 waves / block share the x-tile staged in LDS as f16.
__global__ __launch_bounds__(256) void q4k_wmma_kernel(
    const float* __restrict__ x,      // [64, 4096] fp32
    const int*   __restrict__ packed, // one byte (2 nibbles) per int32
    const float* __restrict__ d,      // [NUM_SUPER]
    const float* __restrict__ dmin,   // [NUM_SUPER]
    const int*   __restrict__ scales, // [NUM_SUPER*8]
    const int*   __restrict__ mins,   // [NUM_SUPER*8]
    float*       __restrict__ dst,    // partials (ws) or final out
    int chunks_per_split, int slice_elems)
{
  __shared__ __align__(64) _Float16 xs[BATCH * XROW];  // 34,816 B

  const int tid  = threadIdx.x;
  const int wave = tid >> 5;
  const int lane = tid & 31;
  const int l16  = lane & 15;
  const int h    = lane >> 4;                   // K-half selector (WMMA layout)
  const int n0   = (blockIdx.x * WAVES_PER_BLOCK + wave) * 16;
  const int o    = n0 + l16;                    // this lane's output column of W

  v8f acc[4] = {};                              // 4 M-tiles of 16 -> M=64

  for (int cc = 0; cc < chunks_per_split; ++cc) {
    const int chunk = blockIdx.y * chunks_per_split + cc;   // 256-wide K chunk

    // ---- stage x[:, chunk*256 : +256] into LDS as f16 ----
    __syncthreads();
    {
      const int c0 = chunk * 256;
      #pragma unroll
      for (int e = 0; e < 16; ++e) {
        const int flat = (e * 256 + tid) * 4;   // 16384 halves, float4-coalesced
        const int row  = flat >> 8;
        const int col  = flat & 255;
        const float4 v = *reinterpret_cast<const float4*>(x + row * IN_DIM + c0 + col);
        _Float16* p = &xs[row * XROW + col];
        p[0] = (_Float16)v.x; p[1] = (_Float16)v.y;
        p[2] = (_Float16)v.z; p[3] = (_Float16)v.w;
      }
    }
    __syncthreads();

    // ---- per-lane super-block constants for this chunk ----
    const int   s  = o * 16 + chunk;            // super index (16 supers per W row)
    const float dv = d[s];
    const float dm = dmin[s];
    const v8i  sc8 = *reinterpret_cast<const v8i*>(scales + s * 8);
    const v8i  mn8 = *reinterpret_cast<const v8i*>(mins   + s * 8);
    const int* pkb = packed + o * (IN_DIM / 2) + chunk * 128 + h * 8;
    __builtin_prefetch(pkb + 128, 0, 1);        // next chunk's weight stream

    #pragma unroll
    for (int step = 0; step < 8; ++step) {      // 8 K-steps of 32
      // w = q*wa + wb ;  via f16 trick: w = (1024+q)*wa + (wb - 1024*wa)
      const float wa  = dv * (float)sc8[step] * (1.0f / 945.0f);
      const float wb  = dv * (float)mn8[step] * (1.0f / 63.0f) + dm;
      const float wbp = __builtin_fmaf(-1024.0f, wa, wb);
      const _Float16 wah  = (_Float16)wa;
      const _Float16 wbph = (_Float16)wbp;
      const v2h wa2  = {wah,  wah};
      const v2h wbp2 = {wbph, wbph};

      // hoist all A-tile LDS loads; their dscnt wait overlaps B dequant VALU
      const int kbase = step * 32;
      v16h A[4];
      #pragma unroll
      for (int mt = 0; mt < 4; ++mt) {
        // A layout: lane half h -> slots 0-15 = K 16h..16h+15 (matches B)
        const int row = mt * 16 + l16;
        const v8h alo = *reinterpret_cast<const v8h*>(&xs[row * XROW + kbase + 16 * h]);
        const v8h ahi = *reinterpret_cast<const v8h*>(&xs[row * XROW + kbase + 16 * h + 8]);
        A[mt] = __builtin_shufflevector(alo, ahi,
                    0,1,2,3,4,5,6,7,8,9,10,11,12,13,14,15);
      }

      // B operand: 16 consecutive K nibbles for column o, k = step*32 + 16*h + t
      const v8i p = *reinterpret_cast<const v8i*>(pkb + step * 16);
      v8i bi;
      #pragma unroll
      for (int u = 0; u < 8; ++u) {
        const int pb = p[u];
        // {lo, hi} nibbles as exact f16 values (1024+q), one packed fma each
        const int t  = ((pb & 15) | ((pb << 12) & 0x000F0000)) | 0x64006400;
        const v2h q2 = __builtin_bit_cast(v2h, t);
        const v2h r2 = q2 * wa2 + wbp2;          // v_pk_fma_f16
        bi[u] = __builtin_bit_cast(int, r2);
      }
      const v16h B = __builtin_bit_cast(v16h, bi);

      #pragma unroll
      for (int mt = 0; mt < 4; ++mt) {
        acc[mt] = __builtin_amdgcn_wmma_f32_16x16x32_f16(
            false, A[mt], false, B, (short)0, acc[mt], false, false);
      }
    }
  }

  // ---- write partial/final: lane (l16,h), vgpr r -> [mt*16 + 8h + r][n0 + l16]
  float* o_ptr = dst + blockIdx.y * slice_elems;
  #pragma unroll
  for (int mt = 0; mt < 4; ++mt) {
    #pragma unroll
    for (int r = 0; r < 8; ++r) {
      o_ptr[(mt * 16 + h * 8 + r) * OUT_DIM + n0 + l16] = acc[mt][r];
    }
  }
}

// Deterministic reduction of KSPLIT partial slices -> out.
__global__ __launch_bounds__(256) void q4k_reduce_kernel(
    const float* __restrict__ ws, float* __restrict__ out)
{
  const int i = (blockIdx.x * 256 + threadIdx.x) * 4;
  float4 a = *reinterpret_cast<const float4*>(ws + i);
  #pragma unroll
  for (int k = 1; k < KSPLIT; ++k) {
    const float4 b = *reinterpret_cast<const float4*>(ws + k * OUT_ELEMS + i);
    a.x += b.x; a.y += b.y; a.z += b.z; a.w += b.w;
  }
  *reinterpret_cast<float4*>(out + i) = a;
}

extern "C" void kernel_launch(void* const* d_in, const int* in_sizes, int n_in,
                              void* d_out, int out_size, void* d_ws, size_t ws_size,
                              hipStream_t stream) {
  (void)in_sizes; (void)n_in; (void)out_size;
  const float* x      = (const float*)d_in[0];
  const int*   packed = (const int*)  d_in[1];
  const float* d      = (const float*)d_in[2];
  const float* dmin   = (const float*)d_in[3];
  const int*   scales = (const int*)  d_in[4];
  const int*   mins   = (const int*)  d_in[5];
  float*       out    = (float*)d_out;

  const size_t need = (size_t)KSPLIT * OUT_ELEMS * sizeof(float);  // 11.3 MB
  if (d_ws != nullptr && ws_size >= need) {
    // K-split x4 for occupancy: 344 blocks / 2752 waves, then reduce.
    float* ws = (float*)d_ws;
    q4k_wmma_kernel<<<dim3(GRID_X, KSPLIT), 256, 0, stream>>>(
        x, packed, d, dmin, scales, mins, ws,
        (IN_DIM / 256) / KSPLIT, OUT_ELEMS);
    q4k_reduce_kernel<<<OUT_ELEMS / (256 * 4), 256, 0, stream>>>(ws, out);
  } else {
    // Fallback: single split writes final output directly.
    q4k_wmma_kernel<<<dim3(GRID_X, 1), 256, 0, stream>>>(
        x, packed, d, dmin, scales, mins, out, IN_DIM / 256, 0);
  }
}
